// GinVirtualReadoutNetwork_82068235092286
// MI455X (gfx1250) — compile-verified, matrix operand order checked by reading, not run on hardware
//
#include <hip/hip_runtime.h>
#include <hip/hip_bf16.h>

// ---------------------------------------------------------------------------
// GinVirtualReadoutNetwork for MI455X (gfx1250).
// fp32 atomic scatter-add for segment sums (exact, L2-resident);
// bf16 WMMA (v_wmma_f32_16x16x32_bf16, fp32 accum) for all MLP GEMMs.
// MLP blocks: 256 threads, 32-row stripe, A-tile staged once in LDS (bf16),
// waves arranged 2(M) x 4(N) -> 4 WMMAs per A-fragment.
// Workspace requirement: ~326 MB.
// ---------------------------------------------------------------------------

typedef __attribute__((ext_vector_type(16))) __bf16 v16bf;
typedef __attribute__((ext_vector_type(8)))  __bf16 v8bf;
typedef __attribute__((ext_vector_type(8)))  float  v8f;

#define DD 256

static constexpr int N_OBS   = 131072;
static constexpr int N_VIRT  = 4096;
static constexpr int N_TASK  = 65536;
static constexpr int N_ACTOR = 65536;
static constexpr int E_GO = 524288, E_GV = 131072, E_OT = 262144, E_VT = 65536, E_TA = 262144;

// ---------------- fragment loader -------------------------------------------
// 16-bit A/B fragment layout (CDNA5 ISA 7.12.2): lane L holds row/col (L%16);
// element e maps to K = kb + (e>=8?16:0) + e%8, where kb = ks*32 + (L>=16?8:0).
// So each fragment is two 8-element contiguous bf16 runs at p and p+16.
__device__ __forceinline__ v16bf frag_from_bf16(const __bf16* p) {
    v8bf lo = *(const v8bf*)p;
    v8bf hi = *(const v8bf*)(p + 16);
    v16bf r;
#pragma unroll
    for (int i = 0; i < 8; ++i) { r[i] = lo[i]; r[8 + i] = hi[i]; }
    return r;
}

// ---------------- fused 2-layer MLP: out = relu(h@W1+b1)@W2+b2 --------------
// Block = 256 threads (8 waves), 32-row stripe, full 256 cols.
// Wave w: M-half mh = (w&1)*16, col group ng = (w>>1)*64 (4 n-tiles of 16).
template <bool FINAL>
__global__ __launch_bounds__(256)
void gin_mlp_kernel(const float* __restrict__ h,
                    const __bf16* __restrict__ W1T, const float* __restrict__ b1,
                    const __bf16* __restrict__ W2T, const float* __restrict__ b2,
                    float* __restrict__ out) {
    __shared__ __align__(16) __bf16 h_s[32 * DD];     // staged A-tile (16 KB)
    __shared__ __align__(16) __bf16 tmp_s[32 * DD];   // relu(h@W1+b1)  (16 KB)

    const int tid  = threadIdx.x;
    const int lane = tid & 31;
    const int w    = tid >> 5;
    const int H    = lane >> 4;        // half-wave (K split)
    const int m    = lane & 15;        // row (A) / col (B) within 16-tile
    const int khalf = H * 8;
    const int row0 = blockIdx.x * 32;
    const int mh   = (w & 1) * 16;     // M-half owned by this wave
    const int ng   = (w >> 1) * 64;    // 64-col group owned by this wave

    // ---- stage h tile: 32 rows x 256 fp32 -> bf16 LDS (cooperative) ----
    {
        const float4* src = (const float4*)(h + (size_t)row0 * DD);
#pragma unroll
        for (int i = 0; i < 8; ++i) {
            const int linear = tid + i * 256;          // float4 index (2048 total)
            const float4 v = src[linear];
            __bf16* d = h_s + linear * 4;
            d[0] = (__bf16)v.x; d[1] = (__bf16)v.y;
            d[2] = (__bf16)v.z; d[3] = (__bf16)v.w;
        }
    }
    __syncthreads();

    // ---- GEMM1: tmp = relu(h @ W1 + b1), K = 256 (8 x k32 steps) ----
    v8f acc[4];
#pragma unroll
    for (int t = 0; t < 4; ++t) {
        const float bb = b1[ng + t * 16 + m];
#pragma unroll
        for (int r = 0; r < 8; ++r) acc[t][r] = bb;
    }
#pragma unroll
    for (int ks = 0; ks < 8; ++ks) {
        const int kb = ks * 32 + khalf;
        v16bf a = frag_from_bf16(h_s + (size_t)(mh + m) * DD + kb);
#pragma unroll
        for (int t = 0; t < 4; ++t) {
            v16bf bt = frag_from_bf16(W1T + (size_t)(ng + t * 16 + m) * DD + kb);
            acc[t] = __builtin_amdgcn_wmma_f32_16x16x32_bf16(false, a, false, bt,
                                                             (short)0, acc[t], false, false);
        }
    }
#pragma unroll
    for (int t = 0; t < 4; ++t) {        // C layout: M = mh + r + 8*H, N = ng + t*16 + m
#pragma unroll
        for (int r = 0; r < 8; ++r) {
            float v = acc[t][r]; v = v > 0.f ? v : 0.f;
            tmp_s[(mh + r + 8 * H) * DD + ng + t * 16 + m] = (__bf16)v;
        }
    }
    __syncthreads();

    // ---- GEMM2 ----
    if (!FINAL) {
#pragma unroll
        for (int t = 0; t < 4; ++t) {
            const float bb = b2[ng + t * 16 + m];
#pragma unroll
            for (int r = 0; r < 8; ++r) acc[t][r] = bb;
        }
#pragma unroll
        for (int ks = 0; ks < 8; ++ks) {
            const int kb = ks * 32 + khalf;
            v16bf a = frag_from_bf16(tmp_s + (size_t)(mh + m) * DD + kb);
#pragma unroll
            for (int t = 0; t < 4; ++t) {
                v16bf bt = frag_from_bf16(W2T + (size_t)(ng + t * 16 + m) * DD + kb);
                acc[t] = __builtin_amdgcn_wmma_f32_16x16x32_bf16(false, a, false, bt,
                                                                 (short)0, acc[t], false, false);
            }
        }
#pragma unroll
        for (int t = 0; t < 4; ++t)
#pragma unroll
            for (int r = 0; r < 8; ++r)
                out[(size_t)(row0 + mh + r + 8 * H) * DD + ng + t * 16 + m] = acc[t][r];
    } else {
        // Final head: W2 zero-padded to 256x16 (W2T: 16x256). Each wave computes
        // its own M-half tile (redundant across ng); waves with ng==0 store col 0.
        v8f c;
        {
            const float bb = b2[m];
#pragma unroll
            for (int r = 0; r < 8; ++r) c[r] = bb;
        }
#pragma unroll
        for (int ks = 0; ks < 8; ++ks) {
            const int kb = ks * 32 + khalf;
            v16bf a = frag_from_bf16(tmp_s + (size_t)(mh + m) * DD + kb);
            v16bf b = frag_from_bf16(W2T + (size_t)m * DD + kb);
            c = __builtin_amdgcn_wmma_f32_16x16x32_bf16(false, a, false, b,
                                                        (short)0, c, false, false);
        }
        if (ng == 0 && m == 0) {
#pragma unroll
            for (int r = 0; r < 8; ++r) out[row0 + mh + r + 8 * H] = c[r];
        }
    }
}

// ---------------- aggregation kernels ---------------------------------------
__global__ __launch_bounds__(256)
void copy_f4_kernel(const float4* __restrict__ src, float4* __restrict__ dst, int n4) {
    int i = blockIdx.x * 256 + threadIdx.x;
    if (i < n4) dst[i] = src[i];
}

// agg[dst[e]] += x_src[src[e]] ; one thread per 4 features, 64 threads/edge
__global__ __launch_bounds__(256)
void scatter_gin_kernel(const float* __restrict__ xsrc, const int* __restrict__ src,
                        const int* __restrict__ dst, float* __restrict__ agg, int E) {
    int idx = blockIdx.x * 256 + threadIdx.x;
    int e = idx >> 6;
    if (e >= E) return;
    int f = (idx & 63) * 4;
    const float4 v = *(const float4*)(xsrc + (size_t)src[e] * DD + f);
    float* o = agg + (size_t)dst[e] * DD + f;
    atomicAdd(o + 0, v.x); atomicAdd(o + 1, v.y);
    atomicAdd(o + 2, v.z); atomicAdd(o + 3, v.w);
}

// GINE: agg[dst[e]] += relu(x_src[src[e]] + eattr[e]*eW + eb)
__global__ __launch_bounds__(256)
void scatter_gine_kernel(const float* __restrict__ xsrc, const int* __restrict__ src,
                         const int* __restrict__ dst, const float* __restrict__ eattr,
                         const float* __restrict__ eW, const float* __restrict__ eb,
                         float* __restrict__ agg, int E) {
    int idx = blockIdx.x * 256 + threadIdx.x;
    int e = idx >> 6;
    if (e >= E) return;
    int f = (idx & 63) * 4;
    const float ea = eattr[e];
    const float4 x  = *(const float4*)(xsrc + (size_t)src[e] * DD + f);
    const float4 wv = *(const float4*)(eW + f);
    const float4 bv = *(const float4*)(eb + f);
    float4 msg;
    msg.x = fmaxf(0.f, x.x + ea * wv.x + bv.x);
    msg.y = fmaxf(0.f, x.y + ea * wv.y + bv.y);
    msg.z = fmaxf(0.f, x.z + ea * wv.z + bv.z);
    msg.w = fmaxf(0.f, x.w + ea * wv.w + bv.w);
    float* o = agg + (size_t)dst[e] * DD + f;
    atomicAdd(o + 0, msg.x); atomicAdd(o + 1, msg.y);
    atomicAdd(o + 2, msg.z); atomicAdd(o + 3, msg.w);
}

// ---------------- weight preparation ----------------------------------------
// WT[n*256+k] = bf16(W[k*256+n])  (W is (din=K, dout=N) row-major)
__global__ __launch_bounds__(256)
void transpose_bf16_kernel(const float* __restrict__ W, __bf16* __restrict__ WT) {
    int n = blockIdx.x, k = threadIdx.x;
    WT[(size_t)n * DD + k] = (__bf16)W[(size_t)k * DD + n];
}

__global__ __launch_bounds__(256)
void build_final_head_kernel(const float* __restrict__ w2, const float* __restrict__ b2,
                             __bf16* __restrict__ w2padT, float* __restrict__ b2pad) {
    int k = threadIdx.x;
#pragma unroll
    for (int n = 0; n < 16; ++n)
        w2padT[(size_t)n * DD + k] = (n == 0) ? (__bf16)w2[k] : (__bf16)0.f;
    if (k < 16) b2pad[k] = (k == 0) ? b2[0] : 0.f;
}

// ---------------------------------------------------------------------------
extern "C" void kernel_launch(void* const* d_in, const int* in_sizes, int n_in,
                              void* d_out, int out_size, void* d_ws, size_t ws_size,
                              hipStream_t stream) {
    (void)in_sizes; (void)n_in; (void)out_size; (void)ws_size;

    const float* x_goal    = (const float*)d_in[0];
    const float* x_obs     = (const float*)d_in[1];
    const float* x_virtual = (const float*)d_in[2];
    const float* x_task    = (const float*)d_in[3];
    const float* x_actor   = (const float*)d_in[4];
    const int *go_src = (const int*)d_in[5],  *go_dst = (const int*)d_in[6];
    const int *gv_src = (const int*)d_in[7],  *gv_dst = (const int*)d_in[8];
    const int *ot_src = (const int*)d_in[9],  *ot_dst = (const int*)d_in[10];
    const int *vt_src = (const int*)d_in[11], *vt_dst = (const int*)d_in[12];
    const int *ta_src = (const int*)d_in[13], *ta_dst = (const int*)d_in[14];
    const float* edge_attr_ot = (const float*)d_in[15];
    const float *ss_W1=(const float*)d_in[16], *ss_b1=(const float*)d_in[17],
                *ss_W2=(const float*)d_in[18], *ss_b2=(const float*)d_in[19];
    const float *sv_W1=(const float*)d_in[20], *sv_b1=(const float*)d_in[21],
                *sv_W2=(const float*)d_in[22], *sv_b2=(const float*)d_in[23];
    const float *st_W1=(const float*)d_in[24], *st_b1=(const float*)d_in[25],
                *st_W2=(const float*)d_in[26], *st_b2=(const float*)d_in[27];
    const float *vt_W1=(const float*)d_in[28], *vt_b1=(const float*)d_in[29],
                *vt_W2=(const float*)d_in[30], *vt_b2=(const float*)d_in[31];
    const float *st_eW=(const float*)d_in[32], *st_eb=(const float*)d_in[33];
    const float *ac_W1=(const float*)d_in[34], *ac_b1=(const float*)d_in[35],
                *ac_W2=(const float*)d_in[36], *ac_b2=(const float*)d_in[37];
    float* logits = (float*)d_out;

    // ---- workspace carve (required: ~326 MB) ----
    char* ws = (char*)d_ws;
    size_t off = 0;
    auto carve = [&](size_t bytes) { char* p = ws + off; off += (bytes + 255) & ~(size_t)255; return p; };
    const size_t WMAT = (size_t)DD * DD * sizeof(__bf16);     // 128 KB
    __bf16* ssW1T = (__bf16*)carve(WMAT);
    __bf16* ssW2T = (__bf16*)carve(WMAT);
    __bf16* svW1T = (__bf16*)carve(WMAT);
    __bf16* svW2T = (__bf16*)carve(WMAT);
    __bf16* stW1T = (__bf16*)carve(WMAT);
    __bf16* stW2T = (__bf16*)carve(WMAT);
    __bf16* vtW1T = (__bf16*)carve(WMAT);
    __bf16* vtW2T = (__bf16*)carve(WMAT);
    __bf16* acW1T = (__bf16*)carve(WMAT);
    __bf16* acW2T = (__bf16*)carve((size_t)16 * DD * sizeof(__bf16));
    float*  b2pad = (float*)carve(16 * sizeof(float));
    float* agg  = (float*)carve((size_t)N_OBS  * DD * sizeof(float)); // 128 MB, reused per layer
    float* x1_1 = (float*)carve((size_t)N_OBS  * DD * sizeof(float)); // 128 MB (later reused as x2_2)
    float* x1_2 = (float*)carve((size_t)N_VIRT * DD * sizeof(float)); // 4 MB
    float* x2_1 = (float*)carve((size_t)N_TASK * DD * sizeof(float)); // 64 MB
    float* x2_2 = x1_1;                                               // x1_1 dead by then

    // ---- prepare weights (bf16, transposed so B-fragments are contiguous) ----
    transpose_bf16_kernel<<<DD, DD, 0, stream>>>(ss_W1, ssW1T);
    transpose_bf16_kernel<<<DD, DD, 0, stream>>>(ss_W2, ssW2T);
    transpose_bf16_kernel<<<DD, DD, 0, stream>>>(sv_W1, svW1T);
    transpose_bf16_kernel<<<DD, DD, 0, stream>>>(sv_W2, svW2T);
    transpose_bf16_kernel<<<DD, DD, 0, stream>>>(st_W1, stW1T);
    transpose_bf16_kernel<<<DD, DD, 0, stream>>>(st_W2, stW2T);
    transpose_bf16_kernel<<<DD, DD, 0, stream>>>(vt_W1, vtW1T);
    transpose_bf16_kernel<<<DD, DD, 0, stream>>>(vt_W2, vtW2T);
    transpose_bf16_kernel<<<DD, DD, 0, stream>>>(ac_W1, acW1T);
    build_final_head_kernel<<<1, DD, 0, stream>>>(ac_W2, ac_b2, acW2T, b2pad);

    auto copyN = [&](const float* s, float* d, int nrows) {
        int n4 = nrows * (DD / 4);
        copy_f4_kernel<<<(n4 + 255) / 256, 256, 0, stream>>>((const float4*)s, (float4*)d, n4);
    };

    // ---- Layer 1: x1_1 = GIN(goal -> obs) ----
    copyN(x_obs, agg, N_OBS);
    scatter_gin_kernel<<<E_GO / 4, 256, 0, stream>>>(x_goal, go_src, go_dst, agg, E_GO);
    gin_mlp_kernel<false><<<N_OBS / 32, 256, 0, stream>>>(agg, ssW1T, ss_b1, ssW2T, ss_b2, x1_1);

    // ---- Layer 2: x1_2 = GIN(goal -> virtual) ----
    copyN(x_virtual, agg, N_VIRT);
    scatter_gin_kernel<<<E_GV / 4, 256, 0, stream>>>(x_goal, gv_src, gv_dst, agg, E_GV);
    gin_mlp_kernel<false><<<N_VIRT / 32, 256, 0, stream>>>(agg, svW1T, sv_b1, svW2T, sv_b2, x1_2);

    // ---- Layer 3: x2_1 = GINE(x1_1 -> task, edge_attr) ----
    copyN(x_task, agg, N_TASK);
    scatter_gine_kernel<<<E_OT / 4, 256, 0, stream>>>(x1_1, ot_src, ot_dst, edge_attr_ot,
                                                      st_eW, st_eb, agg, E_OT);
    gin_mlp_kernel<false><<<N_TASK / 32, 256, 0, stream>>>(agg, stW1T, st_b1, stW2T, st_b2, x2_1);

    // ---- Layer 4: x2_2 = GIN(x1_2 -> x2_1) ----  (x1_1 dead; x2_2 aliases it)
    copyN(x2_1, agg, N_TASK);
    scatter_gin_kernel<<<E_VT / 4, 256, 0, stream>>>(x1_2, vt_src, vt_dst, agg, E_VT);
    gin_mlp_kernel<false><<<N_TASK / 32, 256, 0, stream>>>(agg, vtW1T, vt_b1, vtW2T, vt_b2, x2_2);

    // ---- Layer 5: logits = GIN(x2_2 -> actor), head dim 1 ----
    copyN(x_actor, agg, N_ACTOR);
    scatter_gin_kernel<<<E_TA / 4, 256, 0, stream>>>(x2_2, ta_src, ta_dst, agg, E_TA);
    gin_mlp_kernel<true><<<N_ACTOR / 32, 256, 0, stream>>>(agg, acW1T, ac_b1, acW2T, b2pad, logits);
}
